// ParallelWarper_47863115546756
// MI455X (gfx1250) — matrix operand order repaired and checked by reading, not compiled, not run on hardware
//
#include <hip/hip_runtime.h>
#include <math.h>

typedef __attribute__((ext_vector_type(16))) _Float16 v16h;
typedef __attribute__((ext_vector_type(8)))  _Float16 v8h;
typedef __attribute__((ext_vector_type(8)))  float    v8f;
typedef __attribute__((ext_vector_type(4)))  unsigned int v4u;
typedef __attribute__((ext_vector_type(8)))  int      v8i;
typedef __attribute__((ext_vector_type(4)))  int      v4i;

#define HH 96
#define WW 96
#define WPAD 98
#define HPAD 98
#define NPIX (HH*WW)
#define NPPAD (HPAD*WPAD)

// ---------------------------------------------------------------------------
// TDM: 2D tile load Global -> LDS.  D# bitfields per CDNA5 ISA 8.3/8.4:
// group0: [1:0]=count=1, [63:32]=lds_addr, [120:64]=global_addr, [127:126]=2
// group1: [17:16]=data_size(1=2B), [79:48]=tensor_dim0, [111:80]=tensor_dim1,
//         [127:112]=tile_dim0, [143:128]=tile_dim1, [207:160]=dim0_stride
// Tensor dims set huge so the OOB clamp never engages (tiles are in-bounds
// by construction: activation buffers are pre-padded).
// Toolchain uses the 6-arg builtin: (v4u, v8i, v4i, v4i, v8i, i32 cpol).
// ---------------------------------------------------------------------------
__device__ __forceinline__ void tdm_load_2d(unsigned lds_off, const void* gptr,
                                            unsigned tile_d0, unsigned tile_d1,
                                            unsigned long long stride_elts) {
  unsigned long long ga = (unsigned long long)(uintptr_t)gptr;
  v4u g0;
  g0[0] = 1u;                                    // count=1 (valid), load mode
  g0[1] = lds_off;                               // LDS byte address
  g0[2] = (unsigned)ga;                          // global_addr[31:0]
  g0[3] = (unsigned)(ga >> 32) | (2u << 30);     // global_addr[56:32] | type=2
  const unsigned t0 = 0x40000000u, t1 = 0x40000000u;
  v8i g1;
  g1[0] = (int)(1u << 16);                                         // 2B elems
  g1[1] = (int)((t0 & 0xFFFFu) << 16);                             // dim0 lo
  g1[2] = (int)(((t0 >> 16) & 0xFFFFu) | ((t1 & 0xFFFFu) << 16));  // dim0 hi|dim1 lo
  g1[3] = (int)(((t1 >> 16) & 0xFFFFu) | ((tile_d0 & 0xFFFFu) << 16));
  g1[4] = (int)(tile_d1 & 0xFFFFu);                                // tile_dim1
  g1[5] = (int)(unsigned)(stride_elts & 0xFFFFFFFFull);            // stride lo
  g1[6] = (int)(unsigned)((stride_elts >> 32) & 0xFFFFull);        // stride hi
  g1[7] = 0;
  v4i gz4 = {0, 0, 0, 0};
  v8i gz8 = {0, 0, 0, 0, 0, 0, 0, 0};
  __builtin_amdgcn_tensor_load_to_lds(g0, g1, gz4, gz4, gz8, 0);
}

// ---------------------------------------------------------------------------
__device__ __forceinline__ float samp2d(const float* __restrict__ img, int yi, int xi) {
  bool v = (yi >= 0) & (yi < HH) & (xi >= 0) & (xi < WW);
  int yc = yi < 0 ? 0 : (yi > HH - 1 ? HH - 1 : yi);
  int xc = xi < 0 ? 0 : (xi > WW - 1 ? WW - 1 : xi);
  return v ? img[yc * WW + xc] : 0.0f;
}

// ---------------------------------------------------------------------------
// Stage 1: flow-warp x_src, concat [warped(64), x_cur(64), flow(2), pad(30)]
// into padded-NHWC f16 activation buffer act0[98*98][160] (borders pre-zeroed)
// ---------------------------------------------------------------------------
__global__ void warp_pack_kernel(const float* __restrict__ xsrc,
                                 const float* __restrict__ xcur,
                                 const float* __restrict__ flow,
                                 _Float16* __restrict__ act0) {
  int idx = blockIdx.x * 256 + threadIdx.x;
  if (idx >= 160 * NPIX) return;
  int p = idx % NPIX;
  int c = idx / NPIX;
  int y = p / WW, x = p % WW;
  float v = 0.0f;
  if (c < 64) {
    float px = (float)x + flow[p];          // flow ch0 = x displacement
    float py = (float)y + flow[NPIX + p];   // flow ch1 = y displacement
    float x0f = floorf(px), y0f = floorf(py);
    float wx = px - x0f, wy = py - y0f;
    int x0 = (int)x0f, y0 = (int)y0f;
    const float* img = xsrc + (size_t)c * NPIX;
    v = samp2d(img, y0,     x0    ) * ((1.f - wy) * (1.f - wx))
      + samp2d(img, y0,     x0 + 1) * ((1.f - wy) * wx)
      + samp2d(img, y0 + 1, x0    ) * (wy * (1.f - wx))
      + samp2d(img, y0 + 1, x0 + 1) * (wy * wx);
  } else if (c < 128) {
    v = xcur[(size_t)(c - 64) * NPIX + p];
  } else if (c < 130) {
    v = flow[(size_t)(c - 128) * NPIX + p];
  }
  act0[(size_t)((y + 1) * WPAD + (x + 1)) * 160 + c] = (_Float16)v;
}

// ---------------------------------------------------------------------------
// 3x3 conv as implicit GEMM on v_wmma_f32_16x16x32_f16, TDM-staged A tile.
// One block = one 16-pixel tile; TDM loads the 3x18xCIN activation window to
// LDS once; 4 waves sweep the COUT/16 channel tiles reading A from LDS
// (ds_load) while B (weights, [COUT][9][CIN] tap-major) streams from L2.
// Fragment layouts used here (wave32; contiguous K for A and B):
//  A 16x32 f16: lane L row M=(L&15); elems 0..15 = K kb..kb+15 (kb=(L>>4)*16),
//               B 32x16: lane L col N=(L&15),
//  16 contiguous K at (L>>4)*16.  D: col N=(L&15), VGPR r -> M=r+(L>=16?8:0).
// ---------------------------------------------------------------------------
template<int CIN, int COUT, bool LRELU, bool PAD_OUT>
__global__ void conv3x3_wmma(const _Float16* __restrict__ act,
                             const _Float16* __restrict__ wgt,
                             const float* __restrict__ bias,
                             _Float16* __restrict__ out) {
  const int TN = COUT / 16;
  __shared__ __align__(64) _Float16 atile[3 * 18 * CIN];
  int mt = blockIdx.x;                    // 576 pixel tiles
  int y  = mt / 6;                        // W=96 -> 6 tiles of 16 per row
  int xb = (mt % 6) * 16;
  int wave = threadIdx.x >> 5;
  int lane = threadIdx.x & 31;
  int lh = lane >> 4, ll = lane & 15;
  int kb = lh * 16;

  if (wave == 0) {                        // one TDM op per block
    unsigned lds_off = (unsigned)(uintptr_t)(void*)atile;
    const _Float16* gsrc = act + ((size_t)y * WPAD + xb) * CIN;
    tdm_load_2d(lds_off, gsrc, 18u * CIN, 3u, (unsigned long long)WPAD * CIN);
    __builtin_amdgcn_s_wait_tensorcnt(0);
  }
  __syncthreads();

  for (int nt = wave; nt < TN; nt += 4) {
    int co = nt * 16 + ll;
    const _Float16* wrow = wgt + (size_t)co * (9 * CIN) + lh * 16;
    v8f acc = {};
    for (int tap = 0; tap < 9; ++tap) {
      int ky = tap / 3, kx = tap % 3;
      const _Float16* arow = atile + (size_t)(ky * 18 + (ll + kx)) * CIN + kb;
      const _Float16* wtap = wrow + tap * CIN;
      if (tap + 1 < 9) __builtin_prefetch(wtap + CIN, 0, 1);
#pragma unroll
      for (int c = 0; c < CIN / 32; ++c) {
        v8h alo = *(const v8h*)(arow + c * 32);
        v8h ahi = *(const v8h*)(arow + c * 32 + 8);
        v16h a;
#pragma unroll
        for (int i = 0; i < 8; ++i) { a[i] = alo[i]; a[i + 8] = ahi[i]; }
        v16h b = *(const v16h*)(wtap + c * 32);
        acc = __builtin_amdgcn_wmma_f32_16x16x32_f16(false, a, false, b,
                                                     (short)0, acc, false, false);
      }
    }
    float bv = bias[co];
#pragma unroll
    for (int r = 0; r < 8; ++r) {
      int M = r + lh * 8;
      float val = acc[r] + bv;
      if (LRELU) val = (val >= 0.f) ? val : 0.1f * val;
      if (PAD_OUT)
        out[(size_t)((y + 1) * WPAD + (xb + M) + 1) * COUT + co] = (_Float16)val;
      else
        out[(size_t)(mt * 16 + M) * COUT + co] = (_Float16)val;
    }
  }
}

// ---------------------------------------------------------------------------
// Stage 3: decode offsets (10*tanh + flow_yx) + sigmoid mask, deformable
// bilinear sampling of x_src into samp[9216][576] f16  (K index = ci*9 + kk)
// ---------------------------------------------------------------------------
__global__ void dsample_kernel(const float* __restrict__ xsrc,
                               const float* __restrict__ flow,
                               const _Float16* __restrict__ out4,  // [9216][432]
                               _Float16* __restrict__ samp) {
  int idx = blockIdx.x * 256 + threadIdx.x;
  if (idx >= 144 * NPIX) return;
  int p  = idx % NPIX;
  int gk = idx / NPIX;            // 0..143 = group*9 + tap
  int g = gk / 9, kk = gk % 9;
  int y = p / WW, x = p % WW;
  int ky = kk / 3, kx = kk % 3;
  const _Float16* o = out4 + (size_t)p * 432;
  float dyr = (float)o[g * 18 + kk * 2];
  float dxr = (float)o[g * 18 + kk * 2 + 1];
  float mr  = (float)o[288 + g * 9 + kk];
  float fx = flow[p], fy = flow[NPIX + p];
  float py = 10.0f * tanhf(dyr) + fy + (float)(y - 1 + ky);
  float px = 10.0f * tanhf(dxr) + fx + (float)(x - 1 + kx);
  float m  = 1.0f / (1.0f + expf(-mr));
  float x0f = floorf(px), y0f = floorf(py);
  float wx = px - x0f, wy = py - y0f;
  int x0 = (int)x0f, y0 = (int)y0f;
  float w00 = (1.f - wy) * (1.f - wx), w01 = (1.f - wy) * wx;
  float w10 = wy * (1.f - wx),         w11 = wy * wx;
#pragma unroll
  for (int cc = 0; cc < 4; ++cc) {     // Cg = 64/16 = 4 channels per group
    int ci = g * 4 + cc;
    const float* img = xsrc + (size_t)ci * NPIX;
    float v = samp2d(img, y0, x0) * w00 + samp2d(img, y0, x0 + 1) * w01
            + samp2d(img, y0 + 1, x0) * w10 + samp2d(img, y0 + 1, x0 + 1) * w11;
    samp[(size_t)p * 576 + ci * 9 + kk] = (_Float16)(v * m);
  }
}

// ---------------------------------------------------------------------------
// Stage 4: deformable conv = GEMM [9216 x 64 x 576] on WMMA; the 16x576 samp
// tile is TDM-staged to LDS once per block and shared by the 4 co-tile waves.
// f32 NCHW output (+bias) straight into d_out.
// ---------------------------------------------------------------------------
__global__ void dcn_gemm_wmma(const _Float16* __restrict__ samp,   // [9216][576]
                              const _Float16* __restrict__ wgt,    // [64][576]
                              const float* __restrict__ bias,
                              float* __restrict__ out) {           // [64][9216]
  __shared__ __align__(64) _Float16 stile[16 * 576];
  int mt = blockIdx.x;
  int wave = threadIdx.x >> 5;
  int lane = threadIdx.x & 31;
  int lh = lane >> 4, ll = lane & 15;
  int kb = lh * 16;

  if (wave == 0) {
    unsigned lds_off = (unsigned)(uintptr_t)(void*)stile;
    tdm_load_2d(lds_off, samp + (size_t)mt * 16 * 576, 9216u, 1u, 9216ull);
    __builtin_amdgcn_s_wait_tensorcnt(0);
  }
  __syncthreads();

  int co = wave * 16 + ll;
  const _Float16* wrow = wgt + (size_t)co * 576 + lh * 16;
  const _Float16* arow = stile + (size_t)ll * 576 + kb;
  v8f acc = {};
#pragma unroll
  for (int c = 0; c < 18; ++c) {
    v8h alo = *(const v8h*)(arow + c * 32);
    v8h ahi = *(const v8h*)(arow + c * 32 + 8);
    v16h a;
#pragma unroll
    for (int i = 0; i < 8; ++i) { a[i] = alo[i]; a[i + 8] = ahi[i]; }
    v16h b = *(const v16h*)(wrow + c * 32);
    acc = __builtin_amdgcn_wmma_f32_16x16x32_f16(false, a, false, b,
                                                 (short)0, acc, false, false);
  }
  float bv = bias[co];
#pragma unroll
  for (int r = 0; r < 8; ++r) {
    int M = r + lh * 8;
    out[(size_t)co * NPIX + mt * 16 + M] = acc[r] + bv;
  }
}

// ---------------------------------------------------------------------------
// Weight repacks (once per launch): [COUT][CINS][3][3] f32 -> [COUT][9][CINP] f16
// ---------------------------------------------------------------------------
__global__ void pack_conv_w(const float* __restrict__ src, _Float16* __restrict__ dst,
                            int COUT, int CINS, int CINP) {
  int idx = blockIdx.x * 256 + threadIdx.x;
  if (idx >= COUT * 9 * CINP) return;
  int ci = idx % CINP;
  int rest = idx / CINP;
  int tap = rest % 9;
  int co  = rest / 9;
  float v = (ci < CINS) ? src[((size_t)co * CINS + ci) * 9 + tap] : 0.0f;
  dst[idx] = (_Float16)v;
}

__global__ void cvt_f16(const float* __restrict__ s, _Float16* __restrict__ d, int n) {
  int i = blockIdx.x * 256 + threadIdx.x;
  if (i < n) d[i] = (_Float16)s[i];
}

// ---------------------------------------------------------------------------
extern "C" void kernel_launch(void* const* d_in, const int* in_sizes, int n_in,
                              void* d_out, int out_size, void* d_ws, size_t ws_size,
                              hipStream_t stream) {
  (void)in_sizes; (void)n_in; (void)ws_size;
  const float* x  = (const float*)d_in[0];
  const float* fb = (const float*)d_in[1];
  const float* ff = (const float*)d_in[2];
  const float* w1 = (const float*)d_in[3];
  const float* b1 = (const float*)d_in[4];
  const float* w2 = (const float*)d_in[5];
  const float* b2 = (const float*)d_in[6];
  const float* w3 = (const float*)d_in[7];
  const float* b3 = (const float*)d_in[8];
  const float* w4 = (const float*)d_in[9];
  const float* b4 = (const float*)d_in[10];
  const float* dw = (const float*)d_in[11];
  const float* db = (const float*)d_in[12];
  float* outp = (float*)d_out;

  char* ws = (char*)d_ws;
  size_t off = 0;
  auto alloc = [&](size_t bytes) -> char* {
    char* p = ws + off;
    off = (off + bytes + 255) & ~(size_t)255;
    return p;
  };
  _Float16* w1h  = (_Float16*)alloc((size_t)64  * 9 * 160 * 2);
  _Float16* w2h  = (_Float16*)alloc((size_t)64  * 9 * 64  * 2);
  _Float16* w3h  = (_Float16*)alloc((size_t)64  * 9 * 64  * 2);
  _Float16* w4h  = (_Float16*)alloc((size_t)432 * 9 * 64  * 2);
  _Float16* dwh  = (_Float16*)alloc((size_t)64  * 576 * 2);
  char* actbase = ws + off;
  _Float16* act0 = (_Float16*)alloc((size_t)NPPAD * 160 * 2);
  _Float16* act1 = (_Float16*)alloc((size_t)NPPAD * 64  * 2);
  _Float16* act2 = (_Float16*)alloc((size_t)NPPAD * 64  * 2);
  _Float16* act3 = (_Float16*)alloc((size_t)NPPAD * 64  * 2);
  size_t actbytes = (size_t)(ws + off - actbase);
  _Float16* out4 = (_Float16*)alloc((size_t)NPIX * 432 * 2);
  _Float16* samp = (_Float16*)alloc((size_t)NPIX * 576 * 2);

  // Zero output (covers the two all-zero time slots) and act borders/padding.
  (void)hipMemsetAsync(d_out, 0, (size_t)out_size * sizeof(float), stream);
  (void)hipMemsetAsync(actbase, 0, actbytes, stream);

  // Weight repacks.
  pack_conv_w<<<(64 * 9 * 160 + 255) / 256, 256, 0, stream>>>(w1, w1h, 64, 130, 160);
  pack_conv_w<<<(64 * 9 * 64  + 255) / 256, 256, 0, stream>>>(w2, w2h, 64, 64, 64);
  pack_conv_w<<<(64 * 9 * 64  + 255) / 256, 256, 0, stream>>>(w3, w3h, 64, 64, 64);
  pack_conv_w<<<(432 * 9 * 64 + 255) / 256, 256, 0, stream>>>(w4, w4h, 432, 64, 64);
  cvt_f16<<<(64 * 576 + 255) / 256, 256, 0, stream>>>(dw, dwh, 64 * 576);

  const size_t FR = (size_t)64 * NPIX;   // one frame of one 64-ch tensor
  const size_t FL = (size_t)2 * NPIX;    // one frame of flow

  for (int dir = 0; dir < 2; ++dir) {
    for (int f = 0; f < 6; ++f) {
      const float* xsrc;
      const float* xcur;
      const float* flow;
      float* oslot;
      if (dir == 0) {            // backward: src = x[f+1], cur = x[f]
        xsrc = x + (size_t)(f + 1) * FR;
        xcur = x + (size_t)f * FR;
        flow = fb + (size_t)f * FL;
        oslot = outp + (size_t)f * FR;                  // x_backward[:, f]
      } else {                   // forward: src = x[f], cur = x[f+1]
        xsrc = x + (size_t)f * FR;
        xcur = x + (size_t)(f + 1) * FR;
        flow = ff + (size_t)f * FL;
        oslot = outp + (size_t)(7 + f + 1) * FR;        // x_forward[:, f+1]
      }
      warp_pack_kernel<<<(160 * NPIX + 255) / 256, 256, 0, stream>>>(xsrc, xcur, flow, act0);
      conv3x3_wmma<160, 64, true,  true ><<<576, 128, 0, stream>>>(act0, w1h, b1, act1);
      conv3x3_wmma< 64, 64, true,  true ><<<576, 128, 0, stream>>>(act1, w2h, b2, act2);
      conv3x3_wmma< 64, 64, true,  true ><<<576, 128, 0, stream>>>(act2, w3h, b3, act3);
      conv3x3_wmma< 64, 432, false, false><<<576, 128, 0, stream>>>(act3, w4h, b4, out4);
      dsample_kernel<<<(144 * NPIX + 255) / 256, 256, 0, stream>>>(xsrc, flow, out4, samp);
      dcn_gemm_wmma<<<576, 128, 0, stream>>>(samp, dwh, db, oslot);
    }
  }
}
